// DynamicSnakeConv_82652350644611
// MI455X (gfx1250) — compile-verified, matrix operand order checked by reading, not run on hardware
//
#include <hip/hip_runtime.h>
#include <hip/hip_bf16.h>

typedef __attribute__((ext_vector_type(16))) __bf16 v16bf;
typedef __attribute__((ext_vector_type(8)))  float  v8f;

#define CIN   128
#define OCH   128
#define KRED  1152                 // CIN * 9
#define NPIX  (8 * 64 * 64)        // 32768
#define MTILE 32                   // pixels per workgroup (2 WMMA M-subtiles)
#define XS_ROW 1160                // KRED + 8 pad (bf16 elems) -> 2320B row, bank-conflict free
#define WB_ROW 1152

__device__ __forceinline__ unsigned short f2bf(float f) {
  unsigned int u = __float_as_uint(f);
  u += 0x7FFFu + ((u >> 16) & 1u);           // round-to-nearest-even
  return (unsigned short)(u >> 16);
}

// ---- repack main weights: wb[o][k9*128 + c] = bf16(w_main[o][c][k9]) ----
__global__ __launch_bounds__(256) void wprep(const float* __restrict__ wmain,
                                             unsigned short* __restrict__ wb) {
  int t = blockIdx.x * 256 + threadIdx.x;
  if (t >= OCH * KRED) return;
  int o  = t / KRED;
  int r  = t - o * KRED;
  int k9 = r >> 7;
  int c  = r & 127;
  wb[t] = f2bf(wmain[(o * CIN + c) * 9 + k9]);
}

// ---- offset conv (only the 6 live channels) + cumsum -> pixel-space coords ----
__global__ __launch_bounds__(256) void offconv(const float* __restrict__ x,
                                               const float* __restrict__ woff,
                                               const float* __restrict__ boff,
                                               float* __restrict__ coords) {
  int p = blockIdx.x * 256 + threadIdx.x;
  if (p >= NPIX) return;
  int b = p >> 12;
  int hw = p & 4095;
  int h = hw >> 6;
  int w = hw & 63;
  const float* xb = x + (size_t)b * (CIN * 4096);
  const int chs[6] = {0, 6, 12, 1, 3, 5};   // x-comps (i=0..2,j=0), y-comps (i=0,j=0..2)
  float s[6] = {0.f, 0.f, 0.f, 0.f, 0.f, 0.f};
  for (int kh = 0; kh < 3; ++kh) {
    int ih = h + kh - 1;
    if ((unsigned)ih >= 64u) continue;
    for (int kw = 0; kw < 3; ++kw) {
      int iw = w + kw - 1;
      if ((unsigned)iw >= 64u) continue;
      int q = kh * 3 + kw;
      const float* xp = xb + ih * 64 + iw;
      #pragma unroll 4
      for (int c = 0; c < CIN; ++c) {
        float xv = xp[c * 4096];
        #pragma unroll
        for (int e = 0; e < 6; ++e)
          s[e] = fmaf(xv, woff[(chs[e] * CIN + c) * 9 + q], s[e]);
      }
    }
  }
  float ox0 = s[0] + boff[0];
  float ox1 = ox0 + s[1] + boff[6];
  float ox2 = ox1 + s[2] + boff[12];
  float oy0 = s[3] + boff[1];
  float oy1 = oy0 + s[4] + boff[3];
  float oy2 = oy1 + s[5] + boff[5];
  float xg = -1.0f + (2.0f / 63.0f) * (float)w;
  float yg = -1.0f + (2.0f / 63.0f) * (float)h;
  float* cp = coords + (size_t)p * 6;
  cp[0] = (xg + ox0 + 1.0f) * 32.0f - 0.5f;   // align_corners=False pixel coords
  cp[1] = (xg + ox1 + 1.0f) * 32.0f - 0.5f;
  cp[2] = (xg + ox2 + 1.0f) * 32.0f - 0.5f;
  cp[3] = (yg + oy0 + 1.0f) * 32.0f - 0.5f;
  cp[4] = (yg + oy1 + 1.0f) * 32.0f - 0.5f;
  cp[5] = (yg + oy2 + 1.0f) * 32.0f - 0.5f;
}

// ---- bilinear sample -> LDS bf16 tile -> WMMA GEMM (M=32 pixels per block) ----
__global__ __launch_bounds__(256) void snakemain(const float* __restrict__ x,
                                                 const float* __restrict__ coords,
                                                 const unsigned short* __restrict__ wb,
                                                 const float* __restrict__ bmain,
                                                 float* __restrict__ out) {
  __shared__ unsigned short xsT[MTILE * XS_ROW];   // 74240 B, padded rows
  __shared__ float crd[MTILE * 6];
  int t  = threadIdx.x;
  int p0 = blockIdx.x * MTILE;
  if (t < MTILE * 6) crd[t] = coords[(size_t)p0 * 6 + t];
  __syncthreads();

  // Stage 1: cooperative bilinear sampling into LDS (m = t&31, 16 channels/thread)
  {
    int m  = t & 31;
    int cg = t >> 5;                       // 0..7, 16 channels each
    int p  = p0 + m;
    const float* xb = x + (size_t)(p >> 12) * (CIN * 4096);
    #pragma unroll
    for (int k9 = 0; k9 < 9; ++k9) {
      int i = k9 / 3, j = k9 - i * 3;
      float gx = crd[m * 6 + i];
      float gy = crd[m * 6 + 3 + j];
      float x0f = floorf(gx), y0f = floorf(gy);
      float fx = gx - x0f, fy = gy - y0f;
      int x0 = (int)x0f, y0 = (int)y0f;
      int x1 = x0 + 1, y1 = y0 + 1;
      float vx0 = ((unsigned)x0 < 64u) ? 1.0f : 0.0f;
      float vx1 = ((unsigned)x1 < 64u) ? 1.0f : 0.0f;
      float vy0 = ((unsigned)y0 < 64u) ? 1.0f : 0.0f;
      float vy1 = ((unsigned)y1 < 64u) ? 1.0f : 0.0f;
      float w00 = (1.0f - fx) * (1.0f - fy) * vx0 * vy0;
      float w01 = fx * (1.0f - fy) * vx1 * vy0;
      float w10 = (1.0f - fx) * fy * vx0 * vy1;
      float w11 = fx * fy * vx1 * vy1;
      int xc0 = min(max(x0, 0), 63), xc1 = min(max(x1, 0), 63);
      int yc0 = min(max(y0, 0), 63), yc1 = min(max(y1, 0), 63);
      int a00 = yc0 * 64 + xc0, a01 = yc0 * 64 + xc1;
      int a10 = yc1 * 64 + xc0, a11 = yc1 * 64 + xc1;
      unsigned int pk[8];
      #pragma unroll
      for (int u = 0; u < 8; ++u) {
        const float* c0p = xb + (cg * 16 + u * 2) * 4096;
        const float* c1p = c0p + 4096;
        float v0 = w00 * c0p[a00] + w01 * c0p[a01] + w10 * c0p[a10] + w11 * c0p[a11];
        float v1 = w00 * c1p[a00] + w01 * c1p[a01] + w10 * c1p[a10] + w11 * c1p[a11];
        pk[u] = (unsigned int)f2bf(v0) | ((unsigned int)f2bf(v1) << 16);
      }
      unsigned short* dst = &xsT[m * XS_ROW + k9 * 128 + cg * 16];
      *(uint4*)dst       = make_uint4(pk[0], pk[1], pk[2], pk[3]);
      *(uint4*)(dst + 8) = make_uint4(pk[4], pk[5], pk[6], pk[7]);
    }
  }
  __syncthreads();

  // Stage 2: per-wave 16-wide N tile x two 16-pixel M subtiles;
  //          each B fragment feeds two WMMAs (halves weight traffic)
  int lane = t & 31;
  int wv   = t >> 5;
  int n    = lane & 15;
  int n0   = wv * 16;
  int hoff = (lane < 16) ? 0 : 16;   // bytes: K half selection per ISA fragment layout
  const char* ab0 = (const char*)&xsT[(lane & 15) * XS_ROW];
  const char* ab1 = (const char*)&xsT[((lane & 15) + 16) * XS_ROW];
  const char* bb  = (const char*)(wb + (size_t)(n0 + n) * WB_ROW);
  __builtin_prefetch(bb, 0, 0);
  v8f acc0 = {};
  v8f acc1 = {};
  union Frag { unsigned int u[8]; v16bf v; };
  #pragma unroll 4
  for (int ch = 0; ch < 36; ++ch) {
    int off = ch * 64 + hoff;
    Frag a0, a1, wfr;
    *(uint4*)&wfr.u[0] = *(const uint4*)(bb + off);
    *(uint4*)&wfr.u[4] = *(const uint4*)(bb + off + 32);
    *(uint4*)&a0.u[0]  = *(const uint4*)(ab0 + off);        // K {0..7} / {8..15}
    *(uint4*)&a0.u[4]  = *(const uint4*)(ab0 + off + 32);   // K {16..23} / {24..31}
    *(uint4*)&a1.u[0]  = *(const uint4*)(ab1 + off);
    *(uint4*)&a1.u[4]  = *(const uint4*)(ab1 + off + 32);
    acc0 = __builtin_amdgcn_wmma_f32_16x16x32_bf16(false, a0.v, false, wfr.v,
                                                   (short)0, acc0, false, false);
    acc1 = __builtin_amdgcn_wmma_f32_16x16x32_bf16(false, a1.v, false, wfr.v,
                                                   (short)0, acc1, false, false);
  }
  float bias = bmain[n0 + n];
  int mAdd = (lane < 16) ? 0 : 8;
  int o = n0 + n;
  #pragma unroll
  for (int r = 0; r < 8; ++r) {
    int p = p0 + r + mAdd;
    out[(size_t)(p >> 12) * (OCH * 4096) + (size_t)o * 4096 + (p & 4095)] =
        acc0[r] + bias;
  }
  #pragma unroll
  for (int r = 0; r < 8; ++r) {
    int p = p0 + 16 + r + mAdd;
    out[(size_t)(p >> 12) * (OCH * 4096) + (size_t)o * 4096 + (p & 4095)] =
        acc1[r] + bias;
  }
}

extern "C" void kernel_launch(void* const* d_in, const int* in_sizes, int n_in,
                              void* d_out, int out_size, void* d_ws, size_t ws_size,
                              hipStream_t stream) {
  (void)in_sizes; (void)n_in; (void)out_size; (void)ws_size;
  const float* x     = (const float*)d_in[0];
  const float* woff  = (const float*)d_in[1];
  const float* boff  = (const float*)d_in[2];
  const float* wmain = (const float*)d_in[3];
  const float* bmain = (const float*)d_in[4];
  float* out = (float*)d_out;

  float* coords      = (float*)d_ws;                                    // 32768*6 f32
  unsigned short* wb = (unsigned short*)((char*)d_ws +
                                         (size_t)NPIX * 6 * sizeof(float)); // 128*1152 bf16

  wprep<<<(OCH * KRED + 255) / 256, 256, 0, stream>>>(wmain, wb);
  offconv<<<NPIX / 256, 256, 0, stream>>>(x, woff, boff, coords);
  snakemain<<<NPIX / MTILE, 256, 0, stream>>>(x, coords, wb, bmain, out);
}
